// DARPGREATLayer_77807627534731
// MI455X (gfx1250) — matrix operand order repaired and checked by reading, not compiled, not run on hardware
//
#include <hip/hip_runtime.h>
#include <hip/hip_bf16.h>

#define BB 8
#define NN 160
#define DD 128
#define HH 8
#define FF_ 512
#define ND (NN * DD)      // 20480 floats per (b, row-set)
#define LDA 136           // LDS row stride in halves (272B = 68 dwords -> conflict free)

typedef __attribute__((ext_vector_type(16))) _Float16 v16h;
typedef __attribute__((ext_vector_type(8)))  _Float16 v8h;
typedef __attribute__((ext_vector_type(4)))  _Float16 v4h;
typedef __attribute__((ext_vector_type(8)))  float    v8f;
typedef __attribute__((ext_vector_type(4)))  float    v4f;

// ---------------------------------------------------------------------------
// K0: transpose + convert the 6 D x D projection weights to f16: Wt[n][k] = W[k][n]
// ---------------------------------------------------------------------------
__global__ __launch_bounds__(256) void convert_weights(
    const float* __restrict__ Wq_in, const float* __restrict__ Wk_in,
    const float* __restrict__ Wv_in, const float* __restrict__ Wq_out,
    const float* __restrict__ Wk_out, const float* __restrict__ Wv_out,
    _Float16* __restrict__ Wt6) {
  const float* srcs[6] = {Wq_in, Wk_in, Wv_in, Wq_out, Wk_out, Wv_out};
  int mat = blockIdx.y;
  int idx = blockIdx.x * 256 + threadIdx.x;     // 0 .. 16383  (= n*128 + k)
  int n = idx >> 7, k = idx & 127;
  Wt6[(size_t)mat * DD * DD + idx] = (_Float16)srcs[mat][k * DD + n];
}

// ---------------------------------------------------------------------------
// WMMA fragment loaders (wave32 layouts per CDNA5 ISA 7.12.2)
// ---------------------------------------------------------------------------
// A: 16x32 f16. lane<16: M=lane, K = {k0..k0+7, k0+16..k0+23}
//               lane>=16: M=lane-16, K = {k0+8..k0+15, k0+24..k0+31}
__device__ inline v16h fragA(const _Float16* Xl, int m0, int k0, int lane) {
  const _Float16* p = Xl + (m0 + (lane & 15)) * LDA + k0 + ((lane >> 4) << 3);
  union { v16h v; v8h h[2]; } u;
  u.h[0] = *(const v8h*)p;          // ds_load_b128
  u.h[1] = *(const v8h*)(p + 16);   // ds_load_b128
  return u.v;
}
// B: 32x16 f16 from transposed weights Wt[n][k] (dense, stride 128).
//    lane holds column n = n0 + lane%16, K = {k0+8h..+7, k0+16+8h..+7}, h = lane/16
__device__ inline v16h fragB(const _Float16* Wt, int n0, int k0, int lane) {
  const _Float16* p = Wt + (size_t)(n0 + (lane & 15)) * DD + k0 + ((lane >> 4) << 3);
  union { v16h v; v8h h[2]; } u;
  u.h[0] = *(const v8h*)p;          // global_load_b128
  u.h[1] = *(const v8h*)(p + 16);
  return u.v;
}

// ---------------------------------------------------------------------------
// K1: fused edge attention. grid = (B*N, 2). block = 256 (8 waves, 1 wave/head).
//   side 0: "in"  attention for column j = r (softmax over i, gather stride N*D)
//   side 1: "out" attention for row    i = r (softmax over j, contiguous)
// ---------------------------------------------------------------------------
__global__ __launch_bounds__(256) void edge_attn(
    const float* __restrict__ edge, const _Float16* __restrict__ Wt6,
    const float* __restrict__ bq_in, const float* __restrict__ bk_in,
    const float* __restrict__ bv_in, const float* __restrict__ bq_out,
    const float* __restrict__ bk_out, const float* __restrict__ bv_out,
    float* __restrict__ out_in, float* __restrict__ out_out) {
  __shared__ __align__(16) _Float16 Xl[NN * LDA];   // 43,520 B
  __shared__ __align__(16) float scw[HH * NN];      //  5,120 B
  const int tid = threadIdx.x, lane = tid & 31, hd = tid >> 5;
  const int side = blockIdx.y;
  const int b = blockIdx.x / NN, r = blockIdx.x % NN;
  const size_t bOff = (size_t)b * NN * ND;

  // Stage X (160 x 128) as f16 into LDS, float4 loads / b64 LDS stores.
  for (int idx = tid; idx < (ND >> 2); idx += 256) {
    int rr = idx >> 5;            // 32 float4 per 128-wide row
    int c4 = (idx & 31) << 2;
    size_t g = side ? bOff + (size_t)r * ND + (size_t)rr * DD + c4
                    : bOff + (size_t)rr * ND + (size_t)r * DD + c4;
    const float4 vs = *(const float4*)&edge[g];
    v4h hh;
    hh[0] = (_Float16)vs.x; hh[1] = (_Float16)vs.y;
    hh[2] = (_Float16)vs.z; hh[3] = (_Float16)vs.w;
    *(v4h*)&Xl[rr * LDA + c4] = hh;   // ds_store_b64
  }
  __syncthreads();

  const _Float16* Wq = Wt6 + (size_t)(side * 3 + 0) * DD * DD;
  const _Float16* Wk = Wt6 + (size_t)(side * 3 + 1) * DD * DD;
  const _Float16* Wv = Wt6 + (size_t)(side * 3 + 2) * DD * DD;
  const float* bq = side ? bq_out : bq_in;
  const float* bk = side ? bk_out : bk_in;
  const float* bv = side ? bv_out : bv_in;

  v16h bqf[4], bkf[4];
#pragma unroll
  for (int kk = 0; kk < 4; ++kk) {
    bqf[kk] = fragB(Wq, hd * 16, kk * 32, lane);
    bkf[kk] = fragB(Wk, hd * 16, kk * 32, lane);
  }
  const float qb = bq[hd * 16 + (lane & 15)];
  const float kb = bk[hd * 16 + (lane & 15)];
  float* sc = scw + hd * NN;
  const float scale = 0.25f;  // 1/sqrt(d), d = 16
  const int g3 = lane & 7;
  const int hiHalf = (lane >> 4) << 3;   // 0 for lanes 0-15, 8 for 16-31

  // Q/K pass: per 16-row tile compute Q,K (160x16 slices) and row-dot scores.
  for (int t = 0; t < 10; ++t) {
    v8f qa = {}; v8f ka = {};
#pragma unroll
    for (int kk = 0; kk < 4; ++kk) {
      v16h a = fragA(Xl, t * 16, kk * 32, lane);
      qa = __builtin_amdgcn_wmma_f32_16x16x32_f16(false, a, false, bqf[kk],
                                                  (short)0, qa, false, false);
      ka = __builtin_amdgcn_wmma_f32_16x16x32_f16(false, a, false, bkf[kk],
                                                  (short)0, ka, false, false);
    }
    // Per-lane products for all 8 rows of this half-wave, then a batched
    // butterfly: 8 independent shuffles per level -> 1 dscnt wait per level.
    float p[8];
#pragma unroll
    for (int g = 0; g < 8; ++g) p[g] = (qa[g] + qb) * (ka[g] + kb);
#pragma unroll
    for (int m = 1; m < 16; m <<= 1) {
      float tshf[8];
#pragma unroll
      for (int g = 0; g < 8; ++g) tshf[g] = __shfl_xor(p[g], m);
#pragma unroll
      for (int g = 0; g < 8; ++g) p[g] += tshf[g];
    }
    // every lane now holds all 8 row-sums; lane (group*16 + g), g<8 stores row g
    float e0 = (g3 & 1) ? p[1] : p[0];
    float e1 = (g3 & 1) ? p[3] : p[2];
    float e2 = (g3 & 1) ? p[5] : p[4];
    float e3 = (g3 & 1) ? p[7] : p[6];
    float f0 = (g3 & 2) ? e1 : e0;
    float f1 = (g3 & 2) ? e3 : e2;
    float sv = (g3 & 4) ? f1 : f0;
    if ((lane & 15) < 8)
      sc[t * 16 + hiHalf + g3] = sv * scale;
  }
  __syncthreads();

  // Wave-local softmax over the 160 scores of this head.
  float vals[5];
#pragma unroll
  for (int q5 = 0; q5 < 5; ++q5) vals[q5] = sc[lane + 32 * q5];
  float mx = vals[0];
#pragma unroll
  for (int q5 = 1; q5 < 5; ++q5) mx = fmaxf(mx, vals[q5]);
#pragma unroll
  for (int m = 1; m < 32; m <<= 1) mx = fmaxf(mx, __shfl_xor(mx, m));
  float sum = 0.f;
#pragma unroll
  for (int q5 = 0; q5 < 5; ++q5) { vals[q5] = __expf(vals[q5] - mx); sum += vals[q5]; }
#pragma unroll
  for (int m = 1; m < 32; m <<= 1) sum += __shfl_xor(sum, m);
  float inv = 1.f / sum;
#pragma unroll
  for (int q5 = 0; q5 < 5; ++q5) sc[lane + 32 * q5] = vals[q5] * inv;
  __syncthreads();

  // V pass: recompute V tiles via WMMA, weight by attention probs.
  v16h bvf[4];
#pragma unroll
  for (int kk = 0; kk < 4; ++kk) bvf[kk] = fragB(Wv, hd * 16, kk * 32, lane);
  float acc = 0.f;
  for (int t = 0; t < 10; ++t) {
    v8f va = {};
#pragma unroll
    for (int kk = 0; kk < 4; ++kk) {
      v16h a = fragA(Xl, t * 16, kk * 32, lane);
      va = __builtin_amdgcn_wmma_f32_16x16x32_f16(false, a, false, bvf[kk],
                                                  (short)0, va, false, false);
    }
    // 8 attention probs for this half-wave's rows: two ds_load_b128
    const v4f sA = *(const v4f*)&sc[t * 16 + hiHalf];
    const v4f sB = *(const v4f*)&sc[t * 16 + hiHalf + 4];
#pragma unroll
    for (int g = 0; g < 4; ++g) acc += sA[g] * va[g];
#pragma unroll
    for (int g = 0; g < 4; ++g) acc += sB[g] * va[g + 4];
  }
  acc += __shfl_xor(acc, 16);   // combine the two half-wave partial sums per column
  float* outp = (side ? out_out : out_in) + ((size_t)b * NN + r) * DD + hd * 16;
  if (lane < 16) outp[lane] = acc + bv[hd * 16 + lane];
}

// ---------------------------------------------------------------------------
// K3: node pipeline. one block (128 thr) per (b, n) row.
// ---------------------------------------------------------------------------
__device__ inline float blockReduceSum(float v, float* red) {
#pragma unroll
  for (int m = 1; m < 32; m <<= 1) v += __shfl_xor(v, m);
  __syncthreads();                                // protect red reuse
  if ((threadIdx.x & 31) == 0) red[threadIdx.x >> 5] = v;
  __syncthreads();
  return red[0] + red[1] + red[2] + red[3];
}

__global__ __launch_bounds__(128) void node_update(
    const float* __restrict__ node_emb,
    const float* __restrict__ out_in, const float* __restrict__ out_out,
    const float* __restrict__ Wpi, const float* __restrict__ bpi,
    const float* __restrict__ Wpo, const float* __restrict__ bpo,
    const float* __restrict__ Wf1, const float* __restrict__ bf1,
    const float* __restrict__ Wf2, const float* __restrict__ bf2,
    const float* __restrict__ g_attn, const float* __restrict__ be_attn,
    const float* __restrict__ g_ffn, const float* __restrict__ be_ffn,
    const float* __restrict__ W_edge,
    float* __restrict__ x_out, float* __restrict__ xi_ws, float* __restrict__ xj_ws) {
  __shared__ float oin[DD], oout[DD], x1l[DD], hl[FF_], xl[DD], red[4];
  const int t = threadIdx.x;
  const size_t row = blockIdx.x;                  // b*N + n
  oin[t]  = out_in[row * DD + t];
  oout[t] = out_out[row * DD + t];
  __syncthreads();
  float acc = bpi[t] + bpo[t];
  for (int c = 0; c < DD; ++c)
    acc += oin[c] * Wpi[c * DD + t] + oout[c] * Wpo[c * DD + t];
  float y = node_emb[row * DD + t] + acc;
  float mu  = blockReduceSum(y, red) * (1.f / DD);
  float dv  = y - mu;
  float var = blockReduceSum(dv * dv, red) * (1.f / DD);
  float x1  = dv * rsqrtf(var + 1e-5f) * g_attn[t] + be_attn[t];
  x1l[t] = x1;
  __syncthreads();
#pragma unroll
  for (int p = 0; p < 4; ++p) {
    int f = t + 128 * p;
    float a2 = bf1[f];
    for (int c = 0; c < DD; ++c) a2 += x1l[c] * Wf1[c * FF_ + f];
    hl[f] = fmaxf(a2, 0.f);
  }
  __syncthreads();
  float a3 = bf2[t];
  for (int f = 0; f < FF_; ++f) a3 += hl[f] * Wf2[f * DD + t];
  float y2   = x1 + a3;
  float mu2  = blockReduceSum(y2, red) * (1.f / DD);
  float dv2  = y2 - mu2;
  float var2 = blockReduceSum(dv2 * dv2, red) * (1.f / DD);
  float xv   = dv2 * rsqrtf(var2 + 1e-5f) * g_ffn[t] + be_ffn[t];
  x_out[row * DD + t] = xv;
  xl[t] = xv;
  __syncthreads();
  float xi = 0.f, xj = 0.f;
  for (int c = 0; c < DD; ++c) {
    xi += xl[c] * W_edge[c * DD + t];
    xj += xl[c] * W_edge[(DD + c) * DD + t];
  }
  xi_ws[row * DD + t] = xi;
  xj_ws[row * DD + t] = xj;
}

// ---------------------------------------------------------------------------
// K4: edge_out[b,i,j,:] = xi[b,i,:] + xj[b,j,:] + b_edge. one block per (b,i).
// float4 path: 32 lanes cover a 128-wide channel row, 8 j's in flight.
// ---------------------------------------------------------------------------
__global__ __launch_bounds__(256) void edge_out_kernel(
    const float* __restrict__ xi_ws, const float* __restrict__ xj_ws,
    const float* __restrict__ b_edge, float* __restrict__ eo) {
  const int t = threadIdx.x;
  const int q = t & 31;            // float4 slot within the 128-wide row
  const int jo = t >> 5;           // 8 j's strided across the block
  const int b = blockIdx.x / NN, i = blockIdx.x % NN;
  const float4 xi4 = ((const float4*)xi_ws)[(((size_t)b * NN + i) << 5) + q];
  const float4 be4 = ((const float4*)b_edge)[q];
  float4 base;
  base.x = xi4.x + be4.x; base.y = xi4.y + be4.y;
  base.z = xi4.z + be4.z; base.w = xi4.w + be4.w;
  const float4* xjb = (const float4*)xj_ws + ((size_t)b * NN << 5);
  float4* out = (float4*)eo + ((((size_t)b * NN + i) * NN) << 5);
  for (int j = jo; j < NN; j += 8) {
    const float4 xj4 = xjb[(j << 5) + q];
    float4 v;
    v.x = base.x + xj4.x; v.y = base.y + xj4.y;
    v.z = base.z + xj4.z; v.w = base.w + xj4.w;
    out[((size_t)j << 5) + q] = v;
  }
}

// ---------------------------------------------------------------------------
extern "C" void kernel_launch(void* const* d_in, const int* in_sizes, int n_in,
                              void* d_out, int out_size, void* d_ws, size_t ws_size,
                              hipStream_t stream) {
  const float* node_emb  = (const float*)d_in[0];
  const float* edge_emb  = (const float*)d_in[1];
  const float* W_q_in    = (const float*)d_in[2];   const float* b_q_in  = (const float*)d_in[3];
  const float* W_k_in    = (const float*)d_in[4];   const float* b_k_in  = (const float*)d_in[5];
  const float* W_v_in    = (const float*)d_in[6];   const float* b_v_in  = (const float*)d_in[7];
  const float* W_q_out   = (const float*)d_in[8];   const float* b_q_out = (const float*)d_in[9];
  const float* W_k_out   = (const float*)d_in[10];  const float* b_k_out = (const float*)d_in[11];
  const float* W_v_out   = (const float*)d_in[12];  const float* b_v_out = (const float*)d_in[13];
  const float* W_proj_in = (const float*)d_in[14];  const float* b_proj_in  = (const float*)d_in[15];
  const float* W_proj_out= (const float*)d_in[16];  const float* b_proj_out = (const float*)d_in[17];
  const float* W_f1      = (const float*)d_in[18];  const float* b_f1 = (const float*)d_in[19];
  const float* W_f2      = (const float*)d_in[20];  const float* b_f2 = (const float*)d_in[21];
  const float* g_attn    = (const float*)d_in[22];  const float* be_attn = (const float*)d_in[23];
  const float* g_ffn     = (const float*)d_in[24];  const float* be_ffn  = (const float*)d_in[25];
  const float* W_edge    = (const float*)d_in[26];  const float* b_edge  = (const float*)d_in[27];

  char* ws = (char*)d_ws;
  _Float16* Wt6   = (_Float16*)ws;                              // 6*128*128*2 = 196,608 B
  float* out_in_w = (float*)(ws + 196608);                      // 655,360 B
  float* out_out_w= (float*)(ws + 196608 + 655360);
  float* xi_ws    = (float*)(ws + 196608 + 2 * 655360);
  float* xj_ws    = (float*)(ws + 196608 + 3 * 655360);

  float* x_out = (float*)d_out;                                 // B*N*D
  float* eo    = (float*)d_out + (size_t)BB * NN * DD;          // B*N*N*D

  convert_weights<<<dim3(64, 6), 256, 0, stream>>>(
      W_q_in, W_k_in, W_v_in, W_q_out, W_k_out, W_v_out, Wt6);
  edge_attn<<<dim3(BB * NN, 2), 256, 0, stream>>>(
      edge_emb, Wt6, b_q_in, b_k_in, b_v_in, b_q_out, b_k_out, b_v_out,
      out_in_w, out_out_w);
  node_update<<<dim3(BB * NN), 128, 0, stream>>>(
      node_emb, out_in_w, out_out_w, W_proj_in, b_proj_in, W_proj_out, b_proj_out,
      W_f1, b_f1, W_f2, b_f2, g_attn, be_attn, g_ffn, be_ffn, W_edge,
      x_out, xi_ws, xj_ws);
  edge_out_kernel<<<dim3(BB * NN), 256, 0, stream>>>(xi_ws, xj_ws, b_edge, eo);
}
